// AttnBlock_3513283248816
// MI455X (gfx1250) — compile-verified, matrix operand order checked by reading, not run on hardware
//
#include <hip/hip_runtime.h>
#include <hip/hip_bf16.h>

typedef _Float16 v16h __attribute__((ext_vector_type(16)));
typedef _Float16 v8h  __attribute__((ext_vector_type(8)));
typedef _Float16 v4h  __attribute__((ext_vector_type(4)));
typedef float    v8f  __attribute__((ext_vector_type(8)));

#define BATCH 32
#define HW    1024   // 32*32 pixels per batch
#define C     512
#define GROUPS 32
#define GSIZE 16     // channels per group

// ---------------------------------------------------------------------------
// Fragment loader for V_WMMA_F32_16X16X32_F16 A/B operands.
// 16-bit A-matrix 16x32 layout (ISA 7.12.2): lane L holds row M=L%16,
// half=L/16; VGPRs 0..3 = K[kb+half*8 .. +8), VGPRs 4..7 = K[kb+16+half*8 .. +8).
// Row-major [rows x ld] f16 source, ld % 8 == 0, kb % 32 == 0.
// ---------------------------------------------------------------------------
__device__ __forceinline__ v16h load_frag(const _Float16* __restrict__ base,
                                          int ld, int row, int kb, int half) {
    const _Float16* p = base + (long)row * ld + kb + half * 8;
    union { v16h v; v8h h[2]; } u;
    u.h[0] = *(const v8h*)(p);
    u.h[1] = *(const v8h*)(p + 16);
    return u.v;
}

__device__ __forceinline__ v8f wmma_f16(v16h a, v16h b, v8f c) {
    return __builtin_amdgcn_wmma_f32_16x16x32_f16(
        /*neg_a=*/false, a, /*neg_b=*/false, b,
        /*c_mod=*/(short)0, c, /*reuse_a=*/false, /*reuse_b=*/false);
}

// ---------------------------------------------------------------------------
// Generic batched WMMA GEMM:  C[z] = A[z] (MxK) * B[z] (NxK)^T
// 256 threads = 8 waves arranged 2(M) x 4(N); each wave owns a 32x64 tile
// (2x4 WMMA tiles -> 8 v_wmma per 6 fragment loads per K-step).
// Block tile = 64(M) x 256(N).
// Epilogues:
//   EPI=0 : outH[m*N+n]  = f16(acc + bias[n])            (Q, K)
//   EPI=1 : outH[n*M+m]  = f16(acc + bias[n])            (V, transposed)
//   EPI=2 : outF[m*N+n]  = acc * scale                   (scores)
//   EPI=3 : outH[m*N+n]  = f16(acc)                      (attn @ V)
//   EPI=4 : outF[m*N+n]  = resid[m*N+n] + acc + bias[n]  (proj + residual)
// ---------------------------------------------------------------------------
template<int EPI>
__global__ __launch_bounds__(256)
void gemm_wmma(const _Float16* __restrict__ A, const _Float16* __restrict__ B,
               const float* __restrict__ bias, const float* __restrict__ resid,
               float* __restrict__ outF, _Float16* __restrict__ outH,
               int M, int N, int K, float scale,
               long sA, long sB, long sOut, long sRes)
{
    const int z = blockIdx.z;
    A += (long)z * sA;
    B += (long)z * sB;
    if (EPI == 4) resid += (long)z * sRes;
    if (EPI == 2 || EPI == 4) outF += (long)z * sOut;
    else                      outH += (long)z * sOut;

    const int tid  = threadIdx.x;
    const int lane = tid & 31;
    const int wave = tid >> 5;
    const int wm   = wave & 1;   // 2 waves along M
    const int wn   = wave >> 1;  // 4 waves along N
    const int m0   = blockIdx.y * 64  + wm * 32;
    const int n0   = blockIdx.x * 256 + wn * 64;
    const int mL   = lane & 15;
    const int half = lane >> 4;

    v8f acc[2][4] = {};

    for (int kb = 0; kb < K; kb += 32) {
        if (kb + 32 < K) {  // prefetch next K-slab into near caches
            __builtin_prefetch(A + (long)(m0 + mL) * K + kb + 32 + half * 8, 0, 3);
            if (sB != 0)  // B is per-batch activations (weights stay L2-hot)
                __builtin_prefetch(B + (long)(n0 + mL) * K + kb + 32 + half * 8, 0, 3);
        }
        v16h a0 = load_frag(A, K, m0 + mL,      kb, half);
        v16h a1 = load_frag(A, K, m0 + 16 + mL, kb, half);
        v16h b0 = load_frag(B, K, n0 +      mL, kb, half);
        v16h b1 = load_frag(B, K, n0 + 16 + mL, kb, half);
        v16h b2 = load_frag(B, K, n0 + 32 + mL, kb, half);
        v16h b3 = load_frag(B, K, n0 + 48 + mL, kb, half);
        acc[0][0] = wmma_f16(a0, b0, acc[0][0]);
        acc[0][1] = wmma_f16(a0, b1, acc[0][1]);
        acc[0][2] = wmma_f16(a0, b2, acc[0][2]);
        acc[0][3] = wmma_f16(a0, b3, acc[0][3]);
        acc[1][0] = wmma_f16(a1, b0, acc[1][0]);
        acc[1][1] = wmma_f16(a1, b1, acc[1][1]);
        acc[1][2] = wmma_f16(a1, b2, acc[1][2]);
        acc[1][3] = wmma_f16(a1, b3, acc[1][3]);
    }

    // C/D layout: VGPR r -> M = r + 8*half ; N = lane%16
    #pragma unroll
    for (int i = 0; i < 2; ++i) {
        #pragma unroll
        for (int j = 0; j < 4; ++j) {
            const int nn = n0 + j * 16 + mL;
            #pragma unroll
            for (int r = 0; r < 8; ++r) {
                const int mm = m0 + i * 16 + half * 8 + r;
                const float v = acc[i][j][r];
                if (EPI == 0) {
                    outH[(long)mm * N + nn] = (_Float16)(v + bias[nn]);
                } else if (EPI == 1) {
                    outH[(long)nn * M + mm] = (_Float16)(v + bias[nn]);
                } else if (EPI == 2) {
                    outF[(long)mm * N + nn] = v * scale;
                } else if (EPI == 3) {
                    outH[(long)mm * N + nn] = (_Float16)v;
                } else { // EPI == 4
                    outF[(long)mm * N + nn] = resid[(long)mm * N + nn] + v + bias[nn];
                }
            }
        }
    }
}

// ---------------------------------------------------------------------------
// GroupNorm over (h, w, group-channels) for one (batch, group) per block,
// writing f16 normalized activations. 32*32*16 = 16384 elements per block.
// ---------------------------------------------------------------------------
__global__ __launch_bounds__(256)
void groupnorm_kernel(const float* __restrict__ x,
                      const float* __restrict__ gscale,
                      const float* __restrict__ gbias,
                      _Float16* __restrict__ hn)
{
    const int bg = blockIdx.x;          // 0..1023
    const int b  = bg >> 5;
    const int g  = bg & 31;
    const float*   xp = x  + (long)b * HW * C + g * GSIZE;
    _Float16*      hp = hn + (long)b * HW * C + g * GSIZE;
    const int tid = threadIdx.x;

    float sum = 0.f, sumsq = 0.f;
    for (int idx = tid; idx < HW * GSIZE; idx += 256) {
        const int pix = idx >> 4, ch = idx & 15;
        const float v = xp[(long)pix * C + ch];
        sum += v; sumsq += v * v;
    }
    #pragma unroll
    for (int off = 16; off; off >>= 1) {
        sum   += __shfl_xor(sum,   off, 32);
        sumsq += __shfl_xor(sumsq, off, 32);
    }
    __shared__ float s1[8], s2[8];
    const int lane = tid & 31, wave = tid >> 5;
    if (lane == 0) { s1[wave] = sum; s2[wave] = sumsq; }
    __syncthreads();
    float ts = 0.f, tq = 0.f;
    #pragma unroll
    for (int i = 0; i < 8; ++i) { ts += s1[i]; tq += s2[i]; }
    const float mean = ts * (1.f / (HW * GSIZE));
    const float var  = tq * (1.f / (HW * GSIZE)) - mean * mean;
    const float inv  = rsqrtf(var + 1e-6f);

    for (int idx = tid; idx < HW * GSIZE; idx += 256) {
        const int pix = idx >> 4, ch = idx & 15;
        const float v = xp[(long)pix * C + ch];
        const float r = (v - mean) * inv * gscale[g * GSIZE + ch] + gbias[g * GSIZE + ch];
        hp[(long)pix * C + ch] = (_Float16)r;
    }
}

// ---------------------------------------------------------------------------
// Transpose + f32->f16 conversion of the four 512x512 weight matrices.
// wT[d][c] = (f16) w[c][d], so every GEMM becomes A(MxK) * B(NxK)^T.
// ---------------------------------------------------------------------------
__global__ __launch_bounds__(256)
void convert_weights(const float* __restrict__ wq, const float* __restrict__ wk,
                     const float* __restrict__ wv, const float* __restrict__ wp,
                     _Float16* __restrict__ wqT, _Float16* __restrict__ wkT,
                     _Float16* __restrict__ wvT, _Float16* __restrict__ wpT)
{
    const int idx = blockIdx.x * 256 + threadIdx.x;   // 0 .. 512*512-1
    const int r = idx >> 9;           // input row (c)
    const int d = idx & 511;          // input col (d)
    const long src = (long)r * C + d;
    const long dst = (long)d * C + r;
    wqT[dst] = (_Float16)wq[src];
    wkT[dst] = (_Float16)wk[src];
    wvT[dst] = (_Float16)wv[src];
    wpT[dst] = (_Float16)wp[src];
}

// ---------------------------------------------------------------------------
// Row softmax: one block per row of 1024 f32 scores -> f16 probabilities.
// ---------------------------------------------------------------------------
__global__ __launch_bounds__(256)
void softmax_kernel(const float* __restrict__ S, _Float16* __restrict__ P)
{
    const long row = blockIdx.x;                 // 0 .. 32*1024-1
    const float* srow = S + row * HW;
    _Float16*    prow = P + row * HW;
    const int tid  = threadIdx.x;
    const int lane = tid & 31, wave = tid >> 5;

    const float4 v = ((const float4*)srow)[tid]; // 4 cols per thread

    float m = fmaxf(fmaxf(v.x, v.y), fmaxf(v.z, v.w));
    #pragma unroll
    for (int off = 16; off; off >>= 1) m = fmaxf(m, __shfl_xor(m, off, 32));
    __shared__ float red[8];
    if (lane == 0) red[wave] = m;
    __syncthreads();
    float gm = red[0];
    #pragma unroll
    for (int i = 1; i < 8; ++i) gm = fmaxf(gm, red[i]);
    __syncthreads();

    const float e0 = __expf(v.x - gm);
    const float e1 = __expf(v.y - gm);
    const float e2 = __expf(v.z - gm);
    const float e3 = __expf(v.w - gm);
    float s = e0 + e1 + e2 + e3;
    #pragma unroll
    for (int off = 16; off; off >>= 1) s += __shfl_xor(s, off, 32);
    if (lane == 0) red[wave] = s;
    __syncthreads();
    float gs = 0.f;
    #pragma unroll
    for (int i = 0; i < 8; ++i) gs += red[i];
    const float inv = 1.f / gs;

    v4h o;
    o.x = (_Float16)(e0 * inv);
    o.y = (_Float16)(e1 * inv);
    o.z = (_Float16)(e2 * inv);
    o.w = (_Float16)(e3 * inv);
    ((v4h*)prow)[tid] = o;
}

// ---------------------------------------------------------------------------
// Host-side orchestration.
// ---------------------------------------------------------------------------
extern "C" void kernel_launch(void* const* d_in, const int* in_sizes, int n_in,
                              void* d_out, int out_size, void* d_ws, size_t ws_size,
                              hipStream_t stream)
{
    const float* x  = (const float*)d_in[0];
    const float* ns = (const float*)d_in[1];
    const float* nb = (const float*)d_in[2];
    const float* wq = (const float*)d_in[3];
    const float* bq = (const float*)d_in[4];
    const float* wk = (const float*)d_in[5];
    const float* bk = (const float*)d_in[6];
    const float* wv = (const float*)d_in[7];
    const float* bv = (const float*)d_in[8];
    const float* wp = (const float*)d_in[9];
    const float* bp = (const float*)d_in[10];
    float* out = (float*)d_out;

    // ---- workspace carve-up (bytes) ----
    char* ws = (char*)d_ws;
    const size_t SZ_ACT   = (size_t)BATCH * HW * C * sizeof(_Float16);  // 32 MiB
    const size_t SZ_W     = (size_t)C * C * sizeof(_Float16);           // 512 KiB
    const size_t SZ_SCORE = (size_t)BATCH * HW * HW * sizeof(float);    // 128 MiB
    const size_t SZ_ATTN  = (size_t)BATCH * HW * HW * sizeof(_Float16); // 64 MiB

    _Float16* hn   = (_Float16*)(ws);                 size_t off = SZ_ACT;
    _Float16* wqT  = (_Float16*)(ws + off);           off += SZ_W;
    _Float16* wkT  = (_Float16*)(ws + off);           off += SZ_W;
    _Float16* wvT  = (_Float16*)(ws + off);           off += SZ_W;
    _Float16* wpT  = (_Float16*)(ws + off);           off += SZ_W;
    _Float16* q    = (_Float16*)(ws + off);           off += SZ_ACT;
    _Float16* k    = (_Float16*)(ws + off);           off += SZ_ACT;
    _Float16* vT   = (_Float16*)(ws + off);           off += SZ_ACT;   // [b][c][hw]
    float*    sc   = (float*)   (ws + off);           off += SZ_SCORE;
    _Float16* attn = (_Float16*)(ws + off);           off += SZ_ATTN;
    _Float16* obuf = hn;  // hn is dead after QKV; reuse for attn@V output

    const long AS = (long)HW * C;       // per-batch activation stride (elems)
    const long SS = (long)HW * HW;      // per-batch score stride (elems)
    const float iscale = 1.0f / sqrtf((float)C);

    // 1) weights -> transposed f16
    convert_weights<<<dim3((C * C) / 256), 256, 0, stream>>>(
        wq, wk, wv, wp, wqT, wkT, wvT, wpT);

    // 2) GroupNorm -> hn (f16)
    groupnorm_kernel<<<dim3(BATCH * GROUPS), 256, 0, stream>>>(x, ns, nb, hn);

    // 3) Q, K (row-major f16), V (transposed f16)
    dim3 gQKV(C / 256, HW / 64, BATCH);
    gemm_wmma<0><<<gQKV, 256, 0, stream>>>(hn, wqT, bq, nullptr, nullptr, q,
                                           HW, C, C, 1.f, AS, 0, AS, 0);
    gemm_wmma<0><<<gQKV, 256, 0, stream>>>(hn, wkT, bk, nullptr, nullptr, k,
                                           HW, C, C, 1.f, AS, 0, AS, 0);
    gemm_wmma<1><<<gQKV, 256, 0, stream>>>(hn, wvT, bv, nullptr, nullptr, vT,
                                           HW, C, C, 1.f, AS, 0, AS, 0);

    // 4) scores = Q K^T * (1/sqrt(C))   (f32)
    dim3 gS(HW / 256, HW / 64, BATCH);
    gemm_wmma<2><<<gS, 256, 0, stream>>>(q, k, nullptr, nullptr, sc, nullptr,
                                         HW, HW, C, iscale, AS, AS, SS, 0);

    // 5) softmax rows -> attn (f16)
    softmax_kernel<<<dim3(BATCH * HW), 256, 0, stream>>>(sc, attn);

    // 6) O = attn @ V   (A: [hw x hw] f16, B = vT: [c x hw] f16)
    dim3 gAV(C / 256, HW / 64, BATCH);
    gemm_wmma<3><<<gAV, 256, 0, stream>>>(attn, vT, nullptr, nullptr, nullptr, obuf,
                                          HW, C, HW, 1.f, SS, AS, AS, 0);

    // 7) out = x + O @ Wp + bp   (f32)
    gemm_wmma<4><<<gAV, 256, 0, stream>>>(obuf, wpT, bp, x, out, nullptr,
                                          HW, C, C, 1.f, AS, 0, AS, AS);
}
